// PIB_35759897706598
// MI455X (gfx1250) — matrix operand order repaired and checked by reading, not compiled, not run on hardware
//
#include <hip/hip_runtime.h>
#include <hip/hip_bf16.h>
#include <cstdint>

// ---------------------------------------------------------------------------
// Problem: B=16, T=4096, X=512, H=512, Z=256, C=3, S=128, TOPK=100
// Heavy work: 3 GEMMs (65536x512x512, x512x512, x512x256) -> f16 WMMA, f32 acc
// Tiles move global->LDS via CDNA5 async-to-LDS ops (ASYNCcnt), no VGPR staging.
// ---------------------------------------------------------------------------

typedef __attribute__((ext_vector_type(16))) _Float16 v16h;
typedef __attribute__((ext_vector_type(8)))  _Float16 v8h;
typedef __attribute__((ext_vector_type(8)))  float    v8f;

#define LDSA 40   // 32 + 8 f16 pad: rows 16B aligned, conflict-free frag loads
#define LDSB 40
#define ABUFB (128 * LDSA * 2)   // bytes per A double-buffer slice
#define BBUFB (128 * LDSB * 2)   // bytes per B double-buffer slice

// generic pointer to __shared__ -> LDS byte offset (low 32 bits of the
// aperture-based flat address produced by the addrspacecast)
__device__ __forceinline__ uint32_t lds_off(const void* p) {
    return (uint32_t)(uintptr_t)p;
}

// ---------------------------- f32 -> f16 convert ---------------------------
__global__ void cvt_f32_f16(const float* __restrict__ s, _Float16* __restrict__ d, int n) {
    int i = blockIdx.x * blockDim.x + threadIdx.x;
    int stride = gridDim.x * blockDim.x;
    for (; i < n; i += stride) d[i] = (_Float16)s[i];
}

// convert + transpose weights: W[K=512][N] f32 -> Wt[N][512] f16
// (so GEMM B tiles are K-contiguous per column: plain 16B LDS lines, no scatter)
__global__ void cvt_w_f16_t(const float* __restrict__ w, _Float16* __restrict__ wt, int N) {
    int i = blockIdx.x * blockDim.x + threadIdx.x;
    int stride = gridDim.x * blockDim.x;
    int total = 512 * N;
    for (; i < total; i += stride) {
        int k = i / N, n = i % N;
        wt[(long)n * 512 + k] = (_Float16)w[i];
    }
}

// ------------------------------- WMMA GEMM ---------------------------------
// C[M,N] = A[M,512] * B[512,N] + bias ; optional relu; out f16 or f32.
// Bt is the TRANSPOSED weight (N x 512 f16).
// Block: 256 threads (8 waves), tile 128(M) x 128(N), K-step 32, double buffer.
// Wave tile 32x64 -> 2x4 = 8 WMMAs per K-step per wave.
// A16: A base is 16B aligned (async b128 loads) else async b32 loads.
template <bool RELU, bool F16OUT, bool A16>
__global__ __launch_bounds__(256) void gemm_wmma(
    const _Float16* __restrict__ A, const _Float16* __restrict__ Bt,
    const float* __restrict__ bias, void* __restrict__ outv, int N)
{
    __shared__ _Float16 As[2][128 * LDSA];
    __shared__ _Float16 Bs[2][128 * LDSB];

    const int  t    = threadIdx.x;
    const int  lane = t & 31;
    const int  w    = t >> 5;
    const int  wm   = w >> 1;          // 0..3 : 32-row strip
    const int  wn   = w & 1;           // 0..1 : 64-col strip
    const long m0   = (long)blockIdx.x * 128;
    const int  n0   = blockIdx.y * 128;
    const int  KT   = 512 / 32;

    v8f acc[2][4];
#pragma unroll
    for (int i = 0; i < 2; ++i)
#pragma unroll
        for (int j = 0; j < 4; ++j)
#pragma unroll
            for (int r = 0; r < 8; ++r) acc[i][j][r] = 0.f;

    // A tile: 128 rows x 32 f16 = 512 x 16B -> 2/thread (A16 path)
    const int ar0 = t >> 2,         as0 = (t & 3) * 8;
    const int ar1 = (t + 256) >> 2, as1 = ((t + 256) & 3) * 8;
    // B tile: 128 cols x 32 f16 (column-major via Bt) = 512 x 16B -> 2/thread
    const int bc0 = ar0, bs0 = as0;
    const int bc1 = ar1, bs1 = as1;

    // per-thread LDS destination byte offsets (buffer 0; buffer 1 = +ABUFB/BBUFB)
    const uint32_t la0 = lds_off(&As[0][ar0 * LDSA + as0]);
    const uint32_t la1 = lds_off(&As[0][ar1 * LDSA + as1]);
    const uint32_t lb0 = lds_off(&Bs[0][bc0 * LDSB + bs0]);
    const uint32_t lb1 = lds_off(&Bs[0][bc1 * LDSB + bs1]);
    uint32_t lau[8];
    if constexpr (!A16) {
#pragma unroll
        for (int q = 0; q < 8; ++q) {
            int uid = t + 256 * q;
            int row = uid >> 4, uc = uid & 15;
            lau[q] = lds_off(&As[0][row * LDSA + uc * 2]);
        }
    }

    // CDNA5 async global->LDS tile copy (tracked with ASYNCcnt)
    auto async_fetch = [&](int kt, int buf) {
        if constexpr (A16) {
            asm volatile("global_load_async_to_lds_b128 %0, %1, off"
                         :: "v"(la0 + buf * ABUFB),
                            "v"(A + (m0 + ar0) * 512 + kt * 32 + as0) : "memory");
            asm volatile("global_load_async_to_lds_b128 %0, %1, off"
                         :: "v"(la1 + buf * ABUFB),
                            "v"(A + (m0 + ar1) * 512 + kt * 32 + as1) : "memory");
        } else {
            const unsigned int* pu = (const unsigned int*)A;  // 4B aligned base
#pragma unroll
            for (int q = 0; q < 8; ++q) {
                int uid = t + 256 * q;
                int row = uid >> 4, uc = uid & 15;
                asm volatile("global_load_async_to_lds_b32 %0, %1, off"
                             :: "v"(lau[q] + buf * ABUFB),
                                "v"(pu + ((m0 + row) * 512 + kt * 32) / 2 + uc) : "memory");
            }
        }
        asm volatile("global_load_async_to_lds_b128 %0, %1, off"
                     :: "v"(lb0 + buf * BBUFB),
                        "v"(Bt + (long)(n0 + bc0) * 512 + kt * 32 + bs0) : "memory");
        asm volatile("global_load_async_to_lds_b128 %0, %1, off"
                     :: "v"(lb1 + buf * BBUFB),
                        "v"(Bt + (long)(n0 + bc1) * 512 + kt * 32 + bs1) : "memory");
    };

    async_fetch(0, 0);
    asm volatile("s_wait_asynccnt 0x0" ::: "memory");
    __syncthreads();

    for (int kt = 0; kt < KT; ++kt) {
        const int  cur = kt & 1, nxt = cur ^ 1;
        const bool has = (kt + 1 < KT);
        if (has) async_fetch(kt + 1, nxt);   // fire-and-forget under the WMMAs

        // A frag (16x32 f16): lanes 0-15 -> M=lane, K {0..7,16..23};
        //                     lanes 16-31 -> K {8..15,24..31}   (ISA 7.12.2)
        v16h af[2], bf[4];
#pragma unroll
        for (int i = 0; i < 2; ++i) {
            const _Float16* p =
                &As[cur][(wm * 32 + i * 16 + (lane & 15)) * LDSA + (lane >> 4) * 8];
            v8h lo = *(const v8h*)p;
            v8h hi = *(const v8h*)(p + 16);
            af[i] = __builtin_shufflevector(lo, hi, 0,1,2,3,4,5,6,7,8,9,10,11,12,13,14,15);
        }
        // B frag (32x16 f16): col = lane%16, lanes 0-15 K {0..7,16..23}, 16-31 K {8..15,24..31}
#pragma unroll
        for (int j = 0; j < 4; ++j) {
            const _Float16* p =
                &Bs[cur][(wn * 64 + j * 16 + (lane & 15)) * LDSB + (lane >> 4) * 8];
            v8h lo = *(const v8h*)p;
            v8h hi = *(const v8h*)(p + 16);
            bf[j] = __builtin_shufflevector(lo, hi, 0,1,2,3,4,5,6,7,8,9,10,11,12,13,14,15);
        }
#pragma unroll
        for (int i = 0; i < 2; ++i)
#pragma unroll
            for (int j = 0; j < 4; ++j)
                acc[i][j] = __builtin_amdgcn_wmma_f32_16x16x32_f16(
                    false, af[i], false, bf[j], (short)0, acc[i][j], false, false);

        if (has) asm volatile("s_wait_asynccnt 0x0" ::: "memory");
        __syncthreads();
    }

    // C/D layout: VGPR r -> M = r (lanes 0-15) / r+8 (lanes 16-31), N = lane%16
#pragma unroll
    for (int j = 0; j < 4; ++j) {
        const int col = n0 + wn * 64 + j * 16 + (lane & 15);
        const float bv = bias[col];
#pragma unroll
        for (int i = 0; i < 2; ++i) {
            const long rb = m0 + wm * 32 + i * 16 + ((lane >> 4) << 3);
#pragma unroll
            for (int r = 0; r < 8; ++r) {
                float v = acc[i][j][r] + bv;
                if (RELU) v = fmaxf(v, 0.f);
                long idx = (rb + r) * N + col;
                if (F16OUT) ((_Float16*)outv)[idx] = (_Float16)v;
                else        ((float*)outv)[idx]    = v;
            }
        }
    }
}

// ------------- per-(b,k) sums over T for the axis-1 normalize --------------
// block = (b*32+chunk), 256 threads = 256 columns; 128 rows each; deterministic
__global__ void reduce_z(const float* __restrict__ z,
                         float* __restrict__ p1, float* __restrict__ p2) {
    const int blk = blockIdx.x;           // 0..511 == global_row/128
    const int c   = threadIdx.x;          // 0..255
    const float* base = z + (long)blk * 128 * 256 + c;
    float s1 = 0.f, s2 = 0.f;
    for (int r = 0; r < 128; ++r) {
        float v = base[r * 256];
        s1 += v; s2 += v * v;
    }
    p1[blk * 256 + c] = s1;
    p2[blk * 256 + c] = s2;
}

// zsumn[b,k] = (sum_t z) / max(||z[b,:,k]||, 1e-12)  (fixed-order reduce)
__global__ void reduce_z2(const float* __restrict__ p1, const float* __restrict__ p2,
                          float* __restrict__ zsumn) {
    int i = blockIdx.x * 256 + threadIdx.x;   // 0..4095 ; b=i>>8, k=i&255
    int b = i >> 8, k = i & 255;
    float s1 = 0.f, s2 = 0.f;
    for (int ch = 0; ch < 32; ++ch) {
        s1 += p1[(b * 32 + ch) * 256 + k];
        s2 += p2[(b * 32 + ch) * 256 + k];
    }
    zsumn[i] = s1 / fmaxf(sqrtf(s2), 1e-12f);
}

// mu, softplus(sigma), per-(c,k) proxy norm over S; broadcast mu/sigma outputs
__global__ void proxy_prep(const float* __restrict__ prox, const float* __restrict__ eps,
                           float* __restrict__ mu, float* __restrict__ sg,
                           float* __restrict__ den,
                           float* __restrict__ out_mu, float* __restrict__ out_sg) {
    int t = threadIdx.x;
    for (int i = t; i < 768; i += 256) {        // (c,k)
        int c = i >> 8, k = i & 255;
        float m  = prox[c * 512 + k];
        float sv = prox[c * 512 + 256 + k];
        float s  = sv > 20.f ? sv : log1pf(expf(sv));
        mu[i] = m; sg[i] = s;
        float acc = 0.f;
        for (int ss = 0; ss < 128; ++ss) {
            float zp = m + s * eps[((c * 128) + ss) * 256 + k];
            acc += zp * zp;
        }
        den[i] = fmaxf(sqrtf(acc), 1e-12f);
    }
    // broadcast outputs: (B,C,Z) -> index i, (c,k) = i % 768 (recompute, no deps)
    for (int i = t; i < 12288; i += 256) {
        int ck = i % 768;
        int c = ck >> 8, k = ck & 255;
        float m  = prox[c * 512 + k];
        float sv = prox[c * 512 + 256 + k];
        float s  = sv > 20.f ? sv : log1pf(expf(sv));
        out_mu[i] = m;
        out_sg[i] = s;
    }
}

// att[b,c,s] = (1/T) * sum_k zsumn[b,k] * (mu+sg*eps)/den   (block = b*3+c)
__global__ void att_kernel(const float* __restrict__ zsumn,
                           const float* __restrict__ mu, const float* __restrict__ sg,
                           const float* __restrict__ den, const float* __restrict__ eps,
                           float* __restrict__ att) {
    __shared__ float coefL[256], muL[256], sgL[256];
    const int bc = blockIdx.x, b = bc / 3, c = bc % 3;
    const int t = threadIdx.x;                 // s in 0..127
    for (int k = t; k < 256; k += 128) {
        coefL[k] = zsumn[b * 256 + k] / den[c * 256 + k];
        muL[k]   = mu[c * 256 + k];
        sgL[k]   = sg[c * 256 + k];
    }
    __syncthreads();
    const float* ep = eps + (long)(c * 128 + t) * 256;
    float acc = 0.f;
    for (int k = 0; k < 256; ++k)
        acc += coefL[k] * (muL[k] + sgL[k] * ep[k]);
    att[bc * 128 + t] = acc * (1.f / 4096.f);
}

// top-100 means via stable-rank selection (exact top_k tie semantics), then loss
__global__ void loss_kernel(const float* __restrict__ att, const int* __restrict__ y,
                            float* __restrict__ out_loss) {
    __shared__ float v[256];
    __shared__ float red[256];
    const int t = threadIdx.x;
    float total = 0.f;
    for (int b = 0; b < 16; ++b) {
        const int yb = y[b];
        // ---- positive: att[b, yb, :] (128 values, top 100) ----
        if (t < 128) v[t] = att[(b * 3 + yb) * 128 + t];
        __syncthreads();
        float contrib = 0.f;
        if (t < 128) {
            float xv = v[t]; int rank = 0;
            for (int j = 0; j < 128; ++j) {
                float u = v[j];
                rank += (u > xv) || (u == xv && j < t);
            }
            if (rank < 100) contrib = xv;
        }
        red[t] = contrib;
        __syncthreads();
        for (int s = 128; s > 0; s >>= 1) { if (t < s) red[t] += red[t + s]; __syncthreads(); }
        const float posm = red[0] * 0.01f;
        __syncthreads();
        // ---- negative: two classes != yb, ascending (matches stable argsort) ----
        const int c1 = (yb == 0) ? 1 : 0;
        const int c2 = (yb == 2) ? 1 : 2;
        const int cc = (t < 128) ? c1 : c2;
        v[t] = att[(b * 3 + cc) * 128 + (t & 127)];
        __syncthreads();
        {
            float xv = v[t]; int rank = 0;
            for (int j = 0; j < 256; ++j) {
                float u = v[j];
                rank += (u > xv) || (u == xv && j < t);
            }
            contrib = (rank < 100) ? xv : 0.f;
        }
        red[t] = contrib;
        __syncthreads();
        for (int s = 128; s > 0; s >>= 1) { if (t < s) red[t] += red[t + s]; __syncthreads(); }
        const float negm = red[0] * 0.01f;
        __syncthreads();
        if (t == 0) total += expf(negm - posm);
    }
    if (t == 0) *out_loss = total * (1.f / 16.f);
}

// ------------------------------ workspace map ------------------------------
static const size_t WS_BUF0 = 0;                      // 67108864 B: x16, then h2
static const size_t WS_W1   = 67108864;               // Wt1: 512*512 f16
static const size_t WS_W2   = WS_W1 + 524288;         // Wt2: 512*512 f16
static const size_t WS_W3   = WS_W2 + 524288;         // Wt3: 256*512 f16
static const size_t WS_P1   = WS_W3 + 262144;         // 512*256 f32 partials
static const size_t WS_P2   = WS_P1 + 524288;
static const size_t WS_ZS   = WS_P2 + 524288;         // 4096 f32
static const size_t WS_MU   = WS_ZS + 16384;          // 768 f32
static const size_t WS_SG   = WS_MU + 3072;
static const size_t WS_DEN  = WS_SG + 3072;
static const size_t WS_ATT  = WS_DEN + 3072;          // 6144 f32
// total ~ 69.5 MB

extern "C" void kernel_launch(void* const* d_in, const int* in_sizes, int n_in,
                              void* d_out, int out_size, void* d_ws, size_t ws_size,
                              hipStream_t stream) {
    (void)in_sizes; (void)n_in; (void)out_size; (void)ws_size;
    const float* x    = (const float*)d_in[0];
    const int*   y    = (const int*)  d_in[1];
    const float* W1   = (const float*)d_in[2];
    const float* b1   = (const float*)d_in[3];
    const float* W2   = (const float*)d_in[4];
    const float* b2   = (const float*)d_in[5];
    const float* W3   = (const float*)d_in[6];
    const float* b3   = (const float*)d_in[7];
    const float* prox = (const float*)d_in[8];
    const float* eps  = (const float*)d_in[9];

    float* out = (float*)d_out;
    char*  ws  = (char*)d_ws;

    _Float16* buf0  = (_Float16*)(ws + WS_BUF0);
    _Float16* w1t   = (_Float16*)(ws + WS_W1);
    _Float16* w2t   = (_Float16*)(ws + WS_W2);
    _Float16* w3t   = (_Float16*)(ws + WS_W3);
    float*    p1    = (float*)(ws + WS_P1);
    float*    p2    = (float*)(ws + WS_P2);
    float*    zsumn = (float*)(ws + WS_ZS);
    float*    muw   = (float*)(ws + WS_MU);
    float*    sgw   = (float*)(ws + WS_SG);
    float*    denw  = (float*)(ws + WS_DEN);
    float*    attw  = (float*)(ws + WS_ATT);

    // Output layout: mu_topk[12288] | sigma_topk[12288] | loss[1] | z[16777216]
    float*    zout = out + 24577;
    _Float16* h1   = (_Float16*)zout;   // z region (64 MiB) reused for f16 h1

    // f32 -> f16 conversions (x fits in L2 afterwards: 67 MB < 192 MB);
    // weights are converted AND transposed to N x 512 layout.
    cvt_f32_f16<<<2048, 256, 0, stream>>>(x,  buf0, 16 * 4096 * 512);
    cvt_w_f16_t<<<256,  256, 0, stream>>>(W1, w1t, 512);
    cvt_w_f16_t<<<256,  256, 0, stream>>>(W2, w2t, 512);
    cvt_w_f16_t<<<128,  256, 0, stream>>>(W3, w3t, 256);

    // h1 = relu(x@W1+b1)   A=buf0 (16B aligned), out -> z region (f16)
    gemm_wmma<true,  true,  true ><<<dim3(512, 4), 256, 0, stream>>>(buf0, w1t, b1, (void*)h1,   512);
    // h2 = relu(h1@W2+b2)  A=h1 (z region, only 4B aligned -> async b32 loads)
    gemm_wmma<true,  true,  false><<<dim3(512, 4), 256, 0, stream>>>(h1,   w2t, b2, (void*)buf0, 512);
    // z = h2@W3+b3 (f32)   A=buf0, out -> z region (overwrites dead h1)
    gemm_wmma<false, false, true ><<<dim3(512, 2), 256, 0, stream>>>(buf0, w3t, b3, (void*)zout, 256);

    // per-(b,k) sum / sumsq over T, then normalized T-sum
    reduce_z <<<512, 256, 0, stream>>>(zout, p1, p2);
    reduce_z2<<<16,  256, 0, stream>>>(p1, p2, zsumn);

    // proxies: mu, softplus sigma, per-(c,k) norm; broadcast mu_topk/sigma_topk
    proxy_prep<<<1, 256, 0, stream>>>(prox, eps, muw, sgw, denw, out, out + 12288);

    // att[b,c,s] and the proxy loss
    att_kernel<<<48, 128, 0, stream>>>(zsumn, muw, sgw, denw, eps, attw);
    loss_kernel<<<1, 256, 0, stream>>>(attw, y, out + 24576);
}
